// GDN_38611755991791
// MI455X (gfx1250) — compile-verified, matrix operand order checked by reading, not run on hardware
//
#include <hip/hip_runtime.h>
#include <hip/hip_bf16.h>

// Problem constants (match reference)
#define NN   50000
#define EE   800000
#define INC  256
#define HID  96
#define OUTC 64
#define USE_TDM 1

typedef __bf16 bf16;
typedef __attribute__((ext_vector_type(16))) __bf16 v16bf;
typedef __attribute__((ext_vector_type(8)))  float  v8f;
typedef unsigned int uint32x4 __attribute__((ext_vector_type(4)));
typedef int          int32x8  __attribute__((ext_vector_type(8)));
typedef int          int32x4  __attribute__((ext_vector_type(4)));

#define WMMA_BF16(a, b, c) \
  __builtin_amdgcn_wmma_f32_16x16x32_bf16(false, (a), false, (b), (short)0, (c), false, false)

// ---------------------------------------------------------------------------
// Stage a contiguous packed-B panel (elems bf16) from global into LDS via the
// Tensor Data Mover (1-D D#, 8-byte elements), issued once by wave 0;
// TENSORcnt waited before the block barrier. 6-arg builtin (clang-23 lane).
// ---------------------------------------------------------------------------
__device__ inline void stage_b(const bf16* __restrict__ g, bf16* l, int elems) {
#if USE_TDM && __has_builtin(__builtin_amdgcn_tensor_load_to_lds)
  if (threadIdx.x < 32) {
    const unsigned ne = (unsigned)(elems * 2) >> 3;         // # of 8-byte elements
    unsigned long long ga = (unsigned long long)(size_t)g;  // global byte address
    unsigned lo = (unsigned)(size_t)l;                      // LDS byte address
    uint32x4 g0;
    g0.x = 1u;                                               // count=1 (valid D#)
    g0.y = lo;                                               // lds_addr  (bits 63:32)
    g0.z = (unsigned)ga;                                     // global_addr[31:0]
    g0.w = (unsigned)((ga >> 32) & 0x01FFFFFFu) | (2u << 30);// addr[56:32] | type=2
    int32x8 g1;
    g1[0] = (3 << 16);                                       // data_size=8B, mask=0
    g1[1] = (int)(ne << 16);                                 // tensor_dim0[15:0]
    g1[2] = (int)((ne >> 16) & 0xFFFFu) | (1 << 16);         // tensor_dim0[31:16] | tensor_dim1=1
    g1[3] = (int)(ne << 16);                                 // tile_dim0
    g1[4] = 1;                                               // tile_dim1=1, tile_dim2=0
    g1[5] = (int)ne;                                         // tensor_dim0_stride[31:0]
    g1[6] = 0;                                               // stride hi / dim1_stride
    g1[7] = 0;
    int32x4 z4 = {0, 0, 0, 0};
    int32x8 z8 = {0, 0, 0, 0, 0, 0, 0, 0};
    __builtin_amdgcn_tensor_load_to_lds(g0, g1, z4, z4, z8, 0);
    __builtin_amdgcn_s_wait_tensorcnt(0);
  }
  __syncthreads();
#else
  const uint4* s = (const uint4*)g;
  uint4* d = (uint4*)l;
  const int n16 = elems / 8;                                 // 16-byte chunks
  for (int i = threadIdx.x; i < n16; i += blockDim.x) d[i] = s[i];
  __syncthreads();
#endif
}

// ---------------------------------------------------------------------------
// Operand packing into WMMA fragment layout (CDNA5 16-bit, wave32).
// Packed layout: [tile or (kc,n)][lane 0..31][16 bf16]  -> 32B contiguous/lane.
// ---------------------------------------------------------------------------
// A fragments from f32 row-major [rows x lda]: waves = (NN/16) * kcs
__global__ void k_pack_a(const float* __restrict__ A, int lda, int kcs,
                         bf16* __restrict__ dst, int kc_total, int kc_off) {
  const int w    = (int)((blockIdx.x * blockDim.x + threadIdx.x) >> 5);
  const int lane = threadIdx.x & 31;
  if (w >= (NN / 16) * kcs) return;
  const int tile = w / kcs, kc = w % kcs;
  const int m  = tile * 16 + (lane & 15);
  const int kb = (lane & 16) ? 8 : 0;
  const float* row = A + (long)m * lda + kc * 32 + kb;
  bf16* o = dst + (((long)tile * kc_total + kc_off + kc) * 32 + lane) * 16;
#pragma unroll
  for (int j = 0; j < 8; ++j) {
    const int kk = ((j >= 4) ? 16 : 0) + 2 * (j & 3);
    o[2 * j]     = (bf16)row[kk];
    o[2 * j + 1] = (bf16)row[kk + 1];
  }
}

// fused dithered-quantize + A-pack (kcs = HID/32 = 3): waves = (NN/16)*3
__global__ void k_quant_pack(const float* __restrict__ h, const float* __restrict__ bn,
                             float delta, bf16* __restrict__ dst) {
  const int w    = (int)((blockIdx.x * blockDim.x + threadIdx.x) >> 5);
  const int lane = threadIdx.x & 31;
  if (w >= (NN / 16) * 3) return;
  const int tile = w / 3, kc = w % 3;
  const int m  = tile * 16 + (lane & 15);
  const int kb = (lane & 16) ? 8 : 0;
  const float* row = h  + (long)m * HID + kc * 32 + kb;
  const float* bnr = bn + (long)m * HID + kc * 32 + kb;
  bf16* o = dst + (((long)tile * 3 + kc) * 32 + lane) * 16;
  const float rd = 1.0f / delta;
#pragma unroll
  for (int j = 0; j < 8; ++j) {
    const int kk = ((j >= 4) ? 16 : 0) + 2 * (j & 3);
    float b0 = bnr[kk] * delta, b1 = bnr[kk + 1] * delta;
    o[2 * j]     = (bf16)(floorf((row[kk] + b0) * rd) * delta - b0);
    o[2 * j + 1] = (bf16)(floorf((row[kk + 1] + b1) * rd) * delta - b1);
  }
}

// B fragments from f32 row-major [K x ncols]: waves = kts*nts
__global__ void k_pack_b(const float* __restrict__ W, int ncols, int kts, int nts,
                         bf16* __restrict__ dst) {
  const int w    = (int)((blockIdx.x * blockDim.x + threadIdx.x) >> 5);
  const int lane = threadIdx.x & 31;
  if (w >= kts * nts) return;
  const int kc = w / nts, n = w % nts;
  const int col = n * 16 + (lane & 15);
  const int kb  = (lane & 16) ? 8 : 0;
  bf16* o = dst + (((long)kc * nts + n) * 32 + lane) * 16;
#pragma unroll
  for (int j = 0; j < 8; ++j) {
    const int k = kc * 32 + kb + ((j >= 4) ? 16 : 0) + 2 * (j & 3);
    o[2 * j]     = (bf16)W[(long)k * ncols + col];
    o[2 * j + 1] = (bf16)W[(long)(k + 1) * ncols + col];
  }
}

// ---------------------------------------------------------------------------
// Unified WMMA GEMM: C[NN x NT*16] = Apk @ Bpk (+bias).
// 8 waves/block, TWO 16-row tiles per wave (each LDS B fragment feeds two
// independent WMMAs). Tail handled by wave-uniform pointer clamp: the last
// wave re-reads tile0 for its second A stream (dead acc, store guarded),
// so no EXEC-masked loads appear in the hot loop.
// ---------------------------------------------------------------------------
template <int KT, int NT, bool HAS_BIAS>
__global__ __launch_bounds__(256) void k_gemm(const bf16* __restrict__ Apk,
                                              const bf16* __restrict__ Bpk,
                                              const float* __restrict__ bias,
                                              float* __restrict__ C, int ldc) {
  constexpr int FRAG = 32 * 16;                  // bf16 per fragment (whole wave)
  constexpr int BELEMS = KT * NT * FRAG;
  constexpr int TILES = NN / 16;                 // 3125
  __shared__ __attribute__((aligned(32))) bf16 lds_b[BELEMS];

  stage_b(Bpk, lds_b, BELEMS);                   // TDM -> LDS, then barrier

  const int lane  = threadIdx.x & 31;
  const int tile0 = (blockIdx.x * 8 + (threadIdx.x >> 5)) * 2;
  if (tile0 >= TILES) return;                    // wave-uniform: EXEC all-ones
  const bool two = (tile0 + 1 < TILES);          // wave-uniform tail flag

  v8f acc0[NT] = {}, acc1[NT] = {};
  const v16bf* Af0 = (const v16bf*)(Apk + (long)tile0 * KT * FRAG) + lane;
  const v16bf* Af1 = Af0 + (two ? KT * 32 : 0);  // uniform pointer clamp
  const v16bf* Bf  = (const v16bf*)lds_b + lane;
#pragma unroll
  for (int kc = 0; kc < KT; ++kc) {
    v16bf a0 = Af0[kc * 32];
    v16bf a1 = Af1[kc * 32];
    v16bf b[NT];
#pragma unroll
    for (int n = 0; n < NT; ++n) b[n] = Bf[(kc * NT + n) * 32];
#pragma unroll
    for (int n = 0; n < NT; ++n) {
      acc0[n] = WMMA_BF16(a0, b[n], acc0[n]);
      acc1[n] = WMMA_BF16(a1, b[n], acc1[n]);
    }
  }
  const int nlo = lane & 15;
  const int mo  = (lane & 16) ? 8 : 0;
#pragma unroll
  for (int n = 0; n < NT; ++n) {
    const int col = n * 16 + nlo;
    const float bb = HAS_BIAS ? bias[col] : 0.0f;
#pragma unroll
    for (int i = 0; i < 8; ++i)
      C[(long)(tile0 * 16 + mo + i) * ldc + col] = acc0[n][i] + bb;
    if (two) {
#pragma unroll
      for (int i = 0; i < 8; ++i)
        C[(long)(tile0 * 16 + 16 + mo + i) * ldc + col] = acc1[n][i] + bb;
    }
  }
}

// ---------------- elementwise / graph kernels --------------------------------
__global__ void k_fill1(float* __restrict__ d, int n) {      // self-loop degree
  int i = blockIdx.x * blockDim.x + threadIdx.x;
  if (i < n) d[i] = 1.0f;
}

__global__ void k_degree(const int* __restrict__ dst, float* __restrict__ deg, int e) {
  int i = blockIdx.x * blockDim.x + threadIdx.x;
  if (i < e) atomicAdd(&deg[dst[i]], 1.0f);
}

__global__ void k_rsqrt(float* __restrict__ d, int n) {
  int i = blockIdx.x * blockDim.x + threadIdx.x;
  if (i < n) d[i] = rsqrtf(d[i]);                             // deg >= 1 always
}

// agg initialized with the self-loop contribution m[i]*dis[i]^2
__global__ void k_agginit(const float* __restrict__ m, const float* __restrict__ dis,
                          float* __restrict__ agg) {
  int i = blockIdx.x * blockDim.x + threadIdx.x;
  if (i < NN * HID) {
    float d = dis[i / HID];
    agg[i] = m[i] * d * d;
  }
}

// one wave per edge: 96 features -> 3 atomics per lane
__global__ __launch_bounds__(256) void k_scatter(const int* __restrict__ ei,
                                                 const float* __restrict__ m,
                                                 const float* __restrict__ dis,
                                                 float* __restrict__ agg) {
  const int e    = (int)(blockIdx.x * 8u + (threadIdx.x >> 5));
  const int lane = threadIdx.x & 31;
  if (e >= EE) return;
  const int s = ei[e];
  const int d = ei[EE + e];
  const float norm = dis[s] * dis[d];
  const float* ms = m + (long)s * HID;
  float* ad = agg + (long)d * HID;
#pragma unroll
  for (int c = lane; c < HID; c += 32)
    atomicAdd(&ad[c], ms[c] * norm);
}

__global__ void k_relu(const float* __restrict__ agg, const float* __restrict__ cb,
                       const float* __restrict__ eps, float* __restrict__ h) {
  int i = blockIdx.x * blockDim.x + threadIdx.x;
  if (i < NN * HID) h[i] = fmaxf(agg[i] + cb[i % HID] + eps[i], 0.0f);
}

// ---------------- launch -----------------------------------------------------
extern "C" void kernel_launch(void* const* d_in, const int* in_sizes, int n_in,
                              void* d_out, int out_size, void* d_ws, size_t ws_size,
                              hipStream_t stream) {
  const float* x       = (const float*)d_in[0];
  const int*   ei      = (const int*)  d_in[1];   // [2,E] (JAX default int32)
  const float* proj_w  = (const float*)d_in[2];
  const float* proj_b  = (const float*)d_in[3];
  const float* conv_w  = (const float*)d_in[4];
  const float* conv_b  = (const float*)d_in[5];
  const float* out_w   = (const float*)d_in[6];
  const float* out_b   = (const float*)d_in[7];
  const float* b_noise = (const float*)d_in[8];
  const float* eps     = (const float*)d_in[9];
  float* out = (float*)d_out;

  const int TILES = NN / 16;                       // 3125 (exact)
  const long nh = (long)NN * HID;

  // workspace layout (f32 region, then bf16 packed region)
  float* dis  = (float*)d_ws;
  float* h0   = dis + NN;
  float* h    = h0 + nh;
  float* m    = h  + nh;
  float* agg  = m  + nh;
  bf16* xpk   = (bf16*)(agg + nh);                 // [3125][8][512]
  bf16* hqpk  = xpk  + (long)TILES * 8 * 512;      // [3125][3][512]
  bf16* opk   = hqpk + (long)TILES * 3 * 512;      // [3125][6][512]
  bf16* pwp   = opk  + (long)TILES * 6 * 512;      // [8*6][512]
  bf16* cwp   = pwp  + 8 * 6 * 512;                // 3 x [3*6][512]
  bf16* owp   = cwp  + 3 * 3 * 6 * 512;            // [6*4][512]

  const int TB = 256;
  const int gemm_blocks = ((TILES + 1) / 2 + 7) / 8;   // 1563 wave-pairs -> 196
  auto wgrid = [](long waves) { return (int)((waves * 32 + 255) / 256); };

  // ---- pack weights into fragment layout
  k_pack_b<<<wgrid(8 * 6), TB, 0, stream>>>(proj_w, HID, 8, 6, pwp);
  for (int k = 0; k < 3; ++k)
    k_pack_b<<<wgrid(3 * 6), TB, 0, stream>>>(conv_w + (long)k * HID * HID, HID, 3, 6,
                                              cwp + (long)k * 3 * 6 * 512);
  k_pack_b<<<wgrid(6 * 4), TB, 0, stream>>>(out_w, OUTC, 6, 4, owp);

  // ---- gcn_norm
  k_fill1 <<<(NN + TB - 1) / TB, TB, 0, stream>>>(dis, NN);
  k_degree<<<(EE + TB - 1) / TB, TB, 0, stream>>>(ei + EE, dis, EE);
  k_rsqrt <<<(NN + TB - 1) / TB, TB, 0, stream>>>(dis, NN);

  // ---- projection: h0 = x @ proj_w + proj_b
  k_pack_a<<<wgrid((long)TILES * 8), TB, 0, stream>>>(x, INC, 8, xpk, 8, 0);
  k_gemm<8, 6, true><<<gemm_blocks, TB, 0, stream>>>(xpk, pwp, proj_b, h0, HID);

  // ---- GCN layers
  const float deltas[3] = {1.0f, 0.5f, 0.25f};
  for (int k = 0; k < 3; ++k) {
    const float* hin = (k == 0) ? h0 : h;
    k_quant_pack<<<wgrid((long)TILES * 3), TB, 0, stream>>>(hin, b_noise + (long)k * nh,
                                                            deltas[k], hqpk);
    k_gemm<3, 6, false><<<gemm_blocks, TB, 0, stream>>>(hqpk, cwp + (long)k * 3 * 6 * 512,
                                                        nullptr, m, HID);
    k_agginit<<<(int)((nh + TB - 1) / TB), TB, 0, stream>>>(m, dis, agg);
    k_scatter<<<(EE + 7) / 8, TB, 0, stream>>>(ei, m, dis, agg);
    k_relu<<<(int)((nh + TB - 1) / TB), TB, 0, stream>>>(agg, conv_b + k * HID,
                                                         eps + (long)k * nh, h);
  }

  // ---- output head: out = concat(h0, h) @ out_w + out_b
  k_pack_a<<<wgrid((long)TILES * 3), TB, 0, stream>>>(h0, HID, 3, opk, 6, 0);
  k_pack_a<<<wgrid((long)TILES * 3), TB, 0, stream>>>(h,  HID, 3, opk, 6, 3);
  k_gemm<6, 4, true><<<gemm_blocks, TB, 0, stream>>>(opk, owp, out_b, out, OUTC);
}
